// RetrieverListwiseHardNegLoss_41016937677180
// MI455X (gfx1250) — compile-verified, matrix operand order checked by reading, not run on hardware
//
#include <hip/hip_runtime.h>

// RetrieverListwiseHardNegLoss for MI455X (gfx1250, wave32).
// G=4096 graphs x L=512 edges. One workgroup per graph; deterministic
// two-kernel reduction; flag counting in the finalize kernel runs through
// v_wmma_i32_16x16x64_iu8 (exact integer row-sums with an all-ones B).

#define G_GRAPHS 4096
#define L_EDGES  512
#define TPB      512
#define NWAVES   (TPB / 32)
#define KTOP     32
#define TEMP     0.1f
#define MARGIN_  0.5f
#define NEGBIG   (-1e30f)
#define TINYF    1.17549435e-38f

typedef __attribute__((ext_vector_type(8))) int v8i;

// ---------------- wave / block reductions (wave32) ----------------
__device__ __forceinline__ float waveMax32(float v) {
#pragma unroll
  for (int o = 16; o > 0; o >>= 1) v = fmaxf(v, __shfl_xor(v, o, 32));
  return v;
}
__device__ __forceinline__ float waveSum32(float v) {
#pragma unroll
  for (int o = 16; o > 0; o >>= 1) v += __shfl_xor(v, o, 32);
  return v;
}

__device__ __forceinline__ float blockMax(float v, float* sm) {
  const int lane = threadIdx.x & 31, wid = threadIdx.x >> 5;
  v = waveMax32(v);
  if (lane == 0) sm[wid] = v;
  __syncthreads();
  if (threadIdx.x == 0) {
    float r = sm[0];
#pragma unroll
    for (int i = 1; i < NWAVES; ++i) r = fmaxf(r, sm[i]);
    sm[0] = r;
  }
  __syncthreads();
  float r = sm[0];
  __syncthreads();  // sm reusable afterwards
  return r;
}

__device__ __forceinline__ float blockSum(float v, float* sm) {
  const int lane = threadIdx.x & 31, wid = threadIdx.x >> 5;
  v = waveSum32(v);
  if (lane == 0) sm[wid] = v;
  __syncthreads();
  if (threadIdx.x == 0) {
    float r = sm[0];
#pragma unroll
    for (int i = 1; i < NWAVES; ++i) r += sm[i];
    sm[0] = r;
  }
  __syncthreads();
  float r = sm[0];
  __syncthreads();
  return r;
}

// ---------------- kernel 1: per-graph loss terms ----------------
__global__ __launch_bounds__(TPB) void loss_per_graph(
    const float* __restrict__ logits, const float* __restrict__ targets,
    float* __restrict__ ws_lw, float* __restrict__ ws_ct,
    unsigned char* __restrict__ ws_hp, unsigned char* __restrict__ ws_ac) {
  __shared__ float s_red[NWAVES];
  __shared__ float s_nv[L_EDGES];
  __shared__ float s_hard[KTOP];

  const int g = blockIdx.x;
  const int tid = threadIdx.x;
  const float* lg = logits + (size_t)g * L_EDGES;
  const float* tg = targets + (size_t)g * L_EDGES;

  const float z = lg[tid];
  const bool pos = tg[tid] > 0.5f;
  s_nv[tid] = pos ? NEGBIG : z;  // neg_vals for top-k (visible after 1st barrier)

  const float zs = z / TEMP;  // scaled logits (listwise term)

  const float m_all = blockMax(zs, s_red);
  const float m_pos = blockMax(pos ? zs : NEGBIG, s_red);
  const float cntp  = blockSum(pos ? 1.0f : 0.0f, s_red);
  const float s_all = blockSum(expf(zs - m_all), s_red);
  const float s_pos = blockSum(pos ? expf(zs - m_pos) : 0.0f, s_red);

  // ---- top-32 hard negatives: one wave, register bitonic sort + extraction ----
  if (tid < 32) {
    const int lane = tid;
    float v[16];
#pragma unroll
    for (int j = 0; j < 16; ++j) v[j] = s_nv[lane * 16 + j];

    // descending bitonic sort network on 16 registers (static indices)
#pragma unroll
    for (int k = 2; k <= 16; k <<= 1) {
#pragma unroll
      for (int j = k >> 1; j > 0; j >>= 1) {
#pragma unroll
        for (int i = 0; i < 16; ++i) {
          const int l = i ^ j;
          if (l > i) {
            const bool desc = ((i & k) == 0);
            const float a = v[i], b = v[l];
            const bool sw = desc ? (a < b) : (a > b);
            v[i] = sw ? b : a;
            v[l] = sw ? a : b;
          }
        }
      }
    }

    // 32 rounds: global max of per-lane heads; winner pops its list
#pragma unroll 1
    for (int r = 0; r < KTOP; ++r) {
      const float head = v[0];
      const float m = waveMax32(head);
      const unsigned msk = (unsigned)__ballot(head == m);
      const int winner = __builtin_ctz(msk);
      const bool win = (lane == winner);
      if (lane == 0) s_hard[r] = m;
#pragma unroll
      for (int j = 0; j < 15; ++j) v[j] = win ? v[j + 1] : v[j];
      v[15] = win ? NEGBIG : v[15];
    }
  }
  __syncthreads();

  // ---- pairwise softplus over (positives x top-kg negatives) ----
  const float negc = (float)L_EDGES - cntp;
  const float kgf = fminf(negc, (float)KTOP);
  const int kg = (int)kgf;
  float local = 0.0f;
  if (pos) {
    for (int r = 0; r < kg; ++r) {
      const float x = MARGIN_ + s_hard[r] - z;
      local += fmaxf(x, 0.0f) + log1pf(expf(-fabsf(x)));  // stable softplus
    }
  }
  const float psum = blockSum(local, s_red);

  if (tid == 0) {
    const bool haspos = cntp > 0.0f;
    const float log_den = m_all + logf(fmaxf(s_all, TINYF));
    float lw = 0.0f;
    if (haspos) {
      const float log_num = m_pos + logf(fmaxf(s_pos, TINYF));
      lw = -(log_num - log_den);
    }
    const bool active = haspos && (kg > 0);
    const float contrib = active ? (psum / fmaxf(cntp * kgf, 1.0f)) : 0.0f;
    ws_lw[g] = lw;
    ws_ct[g] = contrib;
    ws_hp[g] = haspos ? 1 : 0;
    ws_ac[g] = active ? 1 : 0;
  }
}

// ---------------- kernel 2: deterministic finalize ----------------
__global__ __launch_bounds__(TPB) void loss_finalize(
    const float* __restrict__ ws_lw, const float* __restrict__ ws_ct,
    const int* __restrict__ ws_hp, const int* __restrict__ ws_ac,
    float* __restrict__ out) {
  __shared__ float red[TPB];
  __shared__ float s_out[2];
  __shared__ int s_cnt[2];
  const int tid = threadIdx.x;

  // fixed-order float sums (bit-deterministic across replays)
  float a = 0.0f;
  for (int i = tid; i < G_GRAPHS; i += TPB) a += ws_lw[i];
  red[tid] = a;
  __syncthreads();
  for (int s = TPB / 2; s > 0; s >>= 1) {
    if (tid < s) red[tid] += red[tid + s];
    __syncthreads();
  }
  if (tid == 0) s_out[0] = red[0];
  __syncthreads();

  a = 0.0f;
  for (int i = tid; i < G_GRAPHS; i += TPB) a += ws_ct[i];
  red[tid] = a;
  __syncthreads();
  for (int s = TPB / 2; s > 0; s >>= 1) {
    if (tid < s) red[tid] += red[tid + s];
    __syncthreads();
  }
  if (tid == 0) s_out[1] = red[0];
  __syncthreads();

  // flag counts via v_wmma_i32_16x16x64_iu8: A = 1024 flag bytes/iter,
  // B = all-ones u8 -> D rows are exact partial sums (layout-agnostic total).
  if (tid < 32) {  // wave 0 only: EXEC all-ones within the wave
    v8i ones;
#pragma unroll
    for (int j = 0; j < 8; ++j) ones[j] = 0x01010101;
    v8i c0 = {}, c1 = {};
#pragma unroll
    for (int it = 0; it < G_GRAPHS / 1024; ++it) {
      const int* p0 = ws_hp + it * 256 + tid * 8;
      const int* p1 = ws_ac + it * 256 + tid * 8;
      v8i a0, a1;
#pragma unroll
      for (int j = 0; j < 8; ++j) {
        a0[j] = p0[j];
        a1[j] = p1[j];
      }
      c0 = __builtin_amdgcn_wmma_i32_16x16x64_iu8(false, a0, false, ones, c0,
                                                  false, false);
      c1 = __builtin_amdgcn_wmma_i32_16x16x64_iu8(false, a1, false, ones, c1,
                                                  false, false);
    }
    // grand total = sum of the 16 distinct row-sums (N=0 lives in lanes 0,16)
    int t0 = 0, t1 = 0;
#pragma unroll
    for (int r = 0; r < 8; ++r) {
      t0 += __shfl(c0[r], 0, 32) + __shfl(c0[r], 16, 32);
      t1 += __shfl(c1[r], 0, 32) + __shfl(c1[r], 16, 32);
    }
    if (tid == 0) {
      s_cnt[0] = t0;
      s_cnt[1] = t1;
    }
  }
  __syncthreads();

  if (tid == 0) {
    const float listwise = s_out[0] / fmaxf((float)s_cnt[0], 1.0f);
    const float pairwise = s_out[1] / fmaxf((float)s_cnt[1], 1.0f);
    out[0] = 1.0f * listwise + 0.5f * pairwise;  // LISTWISE_W, PAIRWISE_W
  }
}

extern "C" void kernel_launch(void* const* d_in, const int* in_sizes, int n_in,
                              void* d_out, int out_size, void* d_ws,
                              size_t ws_size, hipStream_t stream) {
  const float* logits = (const float*)d_in[0];
  const float* targets = (const float*)d_in[1];
  // d_in[2] (edge_batch) unused: segments are equal-sized & sorted.
  // d_in[3] (num_graphs) is a device scalar; grid dims are host-side -> G fixed.
  (void)in_sizes;
  (void)n_in;
  (void)out_size;
  (void)ws_size;

  float* ws_lw = (float*)d_ws;                       // 4096 f32
  float* ws_ct = ws_lw + G_GRAPHS;                   // 4096 f32
  unsigned char* ws_hp = (unsigned char*)(ws_ct + G_GRAPHS);  // 4096 u8
  unsigned char* ws_ac = ws_hp + G_GRAPHS;                    // 4096 u8

  loss_per_graph<<<G_GRAPHS, TPB, 0, stream>>>(logits, targets, ws_lw, ws_ct,
                                               ws_hp, ws_ac);
  loss_finalize<<<1, TPB, 0, stream>>>(ws_lw, ws_ct, (const int*)ws_hp,
                                       (const int*)ws_ac, (float*)d_out);
}
